// LSTMModel_61942018343296
// MI455X (gfx1250) — matrix operand order repaired and checked by reading, not compiled, not run on hardware
//
#include <hip/hip_runtime.h>

#define B_SZ 256
#define T_SZ 512
#define H    128
#define G4   512     // 4*H gates
#define NCLS 10
#define INS  10

typedef unsigned short ushort_t;
typedef __attribute__((ext_vector_type(16))) __bf16 bf16x16;
typedef __attribute__((ext_vector_type(8)))  float  f32x8;

// round-to-nearest-even f32 -> bf16
__device__ __forceinline__ ushort_t f2bf(float f) {
  unsigned u = __float_as_uint(f);
  u += 0x7FFFu + ((u >> 16) & 1u);
  return (ushort_t)(u >> 16);
}

union FragU { uint4 q[2]; bf16x16 v; };

// A fragment (16-bit 16x32): lane reads A[row][kb + (lane>=16?8:0) + 0..7] and +16
__device__ __forceinline__ bf16x16 ld_frag_a(const ushort_t* p) {
  FragU f;
  f.q[0] = *(const uint4*)(p);
  f.q[1] = *(const uint4*)(p + 16);
  return f.v;
}
// B fragment (16-bit 32x16): lane reads W[col][kb + (lane>=16?8:0) + 0..7] and +16
__device__ __forceinline__ bf16x16 ld_frag_b(const ushort_t* p) {
  FragU f;
  f.q[0] = *(const uint4*)(p);
  f.q[1] = *(const uint4*)(p + 16);
  return f.v;
}

__device__ __forceinline__ float sigm(float x) { return 1.0f / (1.0f + expf(-x)); }

// Persistent per-workgroup LSTM scan over a 16-row batch slice.
// A = [h(128) | x(XN) | zero-pad], W = [w_hh | w_ih | 0], LDS-resident.
template <int KSTEPS, int XN>
__global__ __launch_bounds__(256) void lstm_scan(
    const float*    __restrict__ x_f32,  // layer0: [B,T,INS] f32 (else unused)
    const ushort_t* __restrict__ x_bf,   // layer1: [B,T,H] bf16 (else unused)
    const ushort_t* __restrict__ Wcat,   // [G4][KSTEPS*32] bf16 packed weights
    const float*    __restrict__ bias,   // [G4] combined b_ih+b_hh
    ushort_t*       __restrict__ hseq,   // [B,T,H] bf16 out, or null
    float*          __restrict__ hlast)  // [B,H] f32 out at t=T-1, or null
{
  constexpr int LDK = KSTEPS * 32;
  extern __shared__ char smem[];
  ushort_t* sW  = (ushort_t*)smem;             // G4*LDK bf16   (up to 256 KB)
  ushort_t* sA  = sW + G4 * LDK;               // 16*LDK bf16   [h | x | pad]
  float*    sG  = (float*)(sA + 16 * LDK);     // 16*G4  f32 gate staging
  float*    sBi = sG + 16 * G4;                // G4 bias
  float*    sC  = sBi + G4;                    // 16*H cell state

  const int tid = threadIdx.x;
  const int R   = blockIdx.x * 16;             // batch-row base of this workgroup

  // one-time: weights + bias into LDS, zero A (h0=0, pad=0) and cell state
  for (int i = tid; i < (G4 * LDK) / 8; i += 256)
    ((uint4*)sW)[i] = ((const uint4*)Wcat)[i];
  for (int i = tid; i < G4; i += 256) sBi[i] = bias[i];
  for (int i = tid; i < 16 * LDK; i += 256) sA[i] = 0;
  for (int i = tid; i < 16 * H; i += 256) sC[i] = 0.0f;
  __syncthreads();

  const int lane = tid & 31;
  const int wv   = tid >> 5;
  const int arow = lane & 15;
  const int ahk  = (lane >> 4) * 8;    // A-fragment k sub-offset
  const int bko  = (lane >> 4) * 8;    // B-fragment k sub-offset
  const int em   = tid >> 4;           // elementwise: row
  const int ej   = (tid & 15) * 8;     // elementwise: col base (8 cols/thread)

  for (int t = 0; t < T_SZ; ++t) {
    // ---- stage x_t into A cols [H, H+XN) ----
    if constexpr (XN == INS) {
      if (tid < 16 * INS) {
        int m = tid / INS, d = tid - m * INS;
        sA[m * LDK + H + d] = f2bf(x_f32[((size_t)(R + m) * T_SZ + t) * INS + d]);
      }
      if (t + 1 < T_SZ && tid < 16)
        __builtin_prefetch(&x_f32[((size_t)(R + tid) * T_SZ + (t + 1)) * INS], 0, 0);
    } else {
      int base = tid * 8;
      int m = base >> 7, j = base & (H - 1);
      *(uint4*)&sA[m * LDK + H + j] =
          *(const uint4*)&x_bf[((size_t)(R + m) * T_SZ + t) * H + j];
      if (t + 1 < T_SZ && tid < 32)
        __builtin_prefetch(
            &x_bf[((size_t)(R + (tid >> 1)) * T_SZ + (t + 1)) * H + (tid & 1) * 64], 0, 0);
    }
    __syncthreads();

    // ---- gates[16,G4] = A[16,LDK] x Wcat^T : 32 N-tiles over 8 waves ----
    bf16x16 afr[KSTEPS];
#pragma unroll
    for (int kk = 0; kk < KSTEPS; ++kk)
      afr[kk] = ld_frag_a(&sA[arow * LDK + kk * 32 + ahk]);

#pragma unroll
    for (int tt = 0; tt < 4; ++tt) {
      const int NB = (wv * 4 + tt) * 16;
      f32x8 acc = {0.f, 0.f, 0.f, 0.f, 0.f, 0.f, 0.f, 0.f};
#pragma unroll
      for (int kk = 0; kk < KSTEPS; ++kk) {
        bf16x16 bfr = ld_frag_b(&sW[(NB + arow) * LDK + kk * 32 + bko]);
        acc = __builtin_amdgcn_wmma_f32_16x16x32_bf16(
            false, afr[kk], false, bfr, (short)0, acc, false, false);
      }
#pragma unroll
      for (int i = 0; i < 8; ++i)
        sG[(i + ((lane >> 4) << 3)) * G4 + NB + (lane & 15)] = acc[i];
    }
    __syncthreads();

    // ---- elementwise gate math, c/h update (f32) ----
    union { ushort_t s[8]; uint4 q; } hb;
#pragma unroll
    for (int r = 0; r < 8; ++r) {
      int j = ej + r;
      float gi = sigm(sG[em * G4 + j]           + sBi[j]);
      float gf = sigm(sG[em * G4 + H + j]       + sBi[H + j]);
      float gg = tanhf(sG[em * G4 + 2 * H + j]  + sBi[2 * H + j]);
      float go = sigm(sG[em * G4 + 3 * H + j]   + sBi[3 * H + j]);
      float c  = gf * sC[em * H + j] + gi * gg;
      sC[em * H + j] = c;
      float h = go * tanhf(c);
      hb.s[r] = f2bf(h);
      if (hlast && t == T_SZ - 1) hlast[(R + em) * H + j] = h;
    }
    *(uint4*)&sA[em * LDK + ej] = hb.q;   // h feeds next step's A
    if (hseq)
      *(uint4*)&hseq[((size_t)(R + em) * T_SZ + t) * H + ej] = hb.q;
    __syncthreads();
  }
}

// Pack [w_hh | w_ih | pad] -> bf16, combine biases.
__global__ void lstm_prep(const float* __restrict__ w_ih0, const float* __restrict__ w_hh0,
                          const float* __restrict__ b_ih0, const float* __restrict__ b_hh0,
                          const float* __restrict__ w_ih1, const float* __restrict__ w_hh1,
                          const float* __restrict__ b_ih1, const float* __restrict__ b_hh1,
                          ushort_t* __restrict__ W0, ushort_t* __restrict__ W1,
                          float* __restrict__ bias0, float* __restrict__ bias1) {
  int i = blockIdx.x * blockDim.x + threadIdx.x;
  const int N0 = G4 * 160, N1 = G4 * 256;
  if (i < N0) {
    int g = i / 160, k = i - g * 160;
    float v = (k < H) ? w_hh0[g * H + k]
            : (k < H + INS) ? w_ih0[g * INS + (k - H)] : 0.0f;
    W0[i] = f2bf(v);
  } else if (i < N0 + N1) {
    int r = i - N0;
    int g = r / 256, k = r - g * 256;
    float v = (k < H) ? w_hh1[g * H + k] : w_ih1[g * H + (k - H)];
    W1[r] = f2bf(v);
  } else if (i < N0 + N1 + G4) {
    int g = i - N0 - N1;
    bias0[g] = b_ih0[g] + b_hh0[g];
  } else if (i < N0 + N1 + 2 * G4) {
    int g = i - N0 - N1 - G4;
    bias1[g] = b_ih1[g] + b_hh1[g];
  }
}

__global__ void lstm_fc(const float* __restrict__ hlast, const float* __restrict__ fc_w,
                        const float* __restrict__ fc_b, float* __restrict__ out) {
  int i = blockIdx.x * blockDim.x + threadIdx.x;
  if (i >= B_SZ * NCLS) return;
  int b = i / NCLS, c = i - b * NCLS;
  float s = fc_b[c];
  const float* hr = hlast + b * H;
  const float* wr = fc_w + c * H;
#pragma unroll 4
  for (int j = 0; j < H; ++j) s = fmaf(hr[j], wr[j], s);
  out[i] = s;
}

extern "C" void kernel_launch(void* const* d_in, const int* in_sizes, int n_in,
                              void* d_out, int out_size, void* d_ws, size_t ws_size,
                              hipStream_t stream) {
  const float* x     = (const float*)d_in[0];
  const float* w_ih0 = (const float*)d_in[1];
  const float* w_hh0 = (const float*)d_in[2];
  const float* b_ih0 = (const float*)d_in[3];
  const float* b_hh0 = (const float*)d_in[4];
  const float* w_ih1 = (const float*)d_in[5];
  const float* w_hh1 = (const float*)d_in[6];
  const float* b_ih1 = (const float*)d_in[7];
  const float* b_hh1 = (const float*)d_in[8];
  const float* fc_w  = (const float*)d_in[9];
  const float* fc_b  = (const float*)d_in[10];
  float* out = (float*)d_out;

  // workspace layout (16B-aligned offsets); total ~34.1 MB
  char* ws = (char*)d_ws;
  ushort_t* W0    = (ushort_t*)(ws + 0);                   // 512*160*2 = 163840
  ushort_t* W1    = (ushort_t*)(ws + 163840);              // 512*256*2 = 262144
  float*    bias0 = (float*)(ws + 425984);                 // 2048
  float*    bias1 = (float*)(ws + 428032);                 // 2048
  ushort_t* h1seq = (ushort_t*)(ws + 430080);              // 256*512*128*2 = 33554432
  float*    hlast = (float*)(ws + 430080 + 33554432);      // 256*128*4 = 131072

  const int prepN = G4 * 160 + G4 * 256 + 2 * G4;
  lstm_prep<<<(prepN + 255) / 256, 256, 0, stream>>>(
      w_ih0, w_hh0, b_ih0, b_hh0, w_ih1, w_hh1, b_ih1, b_hh1, W0, W1, bias0, bias1);

  const size_t smem0 = (size_t)G4 * 160 * 2 + 16 * 160 * 2 + 16 * G4 * 4 + G4 * 4 + 16 * H * 4; // ~207 KB
  const size_t smem1 = (size_t)G4 * 256 * 2 + 16 * 256 * 2 + 16 * G4 * 4 + G4 * 4 + 16 * H * 4; // ~306 KB
  hipFuncSetAttribute(reinterpret_cast<const void*>(&lstm_scan<5, INS>),
                      hipFuncAttributeMaxDynamicSharedMemorySize, (int)smem0);
  hipFuncSetAttribute(reinterpret_cast<const void*>(&lstm_scan<8, H>),
                      hipFuncAttributeMaxDynamicSharedMemorySize, (int)smem1);

  // layer 0: K = 128(h) + 10(x) + 22(pad) = 160 -> 5 k-steps
  lstm_scan<5, INS><<<16, 256, smem0, stream>>>(x, nullptr, W0, bias0, h1seq, nullptr);
  // layer 1: K = 128(h) + 128(x=h1) = 256 -> 8 k-steps
  lstm_scan<8, H><<<16, 256, smem1, stream>>>(nullptr, h1seq, W1, bias1, nullptr, hlast);

  lstm_fc<<<(B_SZ * NCLS + 255) / 256, 256, 0, stream>>>(hlast, fc_w, fc_b, out);
}
